// VeduBitConv2d_44135083934121
// MI455X (gfx1250) — matrix operand order repaired and checked by reading, not compiled, not run on hardware
//
#include <hip/hip_runtime.h>
#include <hip/hip_bf16.h>

// ---- types for WMMA ----
typedef __attribute__((ext_vector_type(16))) _Float16 v16h;
typedef __attribute__((ext_vector_type(8)))  _Float16 v8h;
typedef __attribute__((ext_vector_type(8)))  float    v8f;

#define CIN   64
#define COUT  128
#define H_    112
#define W_    112
#define HW    12544      // 112*112
#define KTOT  576        // 9 taps * 64 channels, reordered k' = off*64 + ci
#define NPIX  401408     // 32*HW
#define NW    73728      // COUT*CIN*3*3
#define NKS   18         // 576/32 K-steps
#define NBLK  128        // pixels per block (12544 % 128 == 0)

// ---- async global->LDS (gfx1250) with sync fallback ----
#if defined(__has_builtin)
#  if __has_builtin(__builtin_amdgcn_global_load_async_to_lds_b128)
#    define HAVE_ASYNC_LDS 1
#  endif
#endif
#ifndef HAVE_ASYNC_LDS
#  define HAVE_ASYNC_LDS 0
#endif

#if HAVE_ASYNC_LDS
typedef __attribute__((ext_vector_type(4))) int v4i;
typedef __attribute__((address_space(1))) v4i as1_v4i;   // global int4
typedef __attribute__((address_space(3))) v4i as3_v4i;   // LDS int4
#endif

__device__ __forceinline__ void wait_async_lds() {
#if HAVE_ASYNC_LDS
#  if __has_builtin(__builtin_amdgcn_s_wait_asynccnt)
    __builtin_amdgcn_s_wait_asynccnt(0);
#  else
    asm volatile("s_wait_asynccnt 0x0" ::: "memory");
#  endif
#endif
}

// ---------------------------------------------------------------------------
// Kernel 1: per-sample mean / rstd over (C,H,W).  One block per sample.
// ---------------------------------------------------------------------------
__global__ void stats_kernel(const float* __restrict__ x, float* __restrict__ stats) {
    const int n   = blockIdx.x;
    const int tid = threadIdx.x;
    const float4* xp = (const float4*)(x + (long)n * CIN * HW);
    const int total4 = (CIN * HW) / 4;   // 200704

    float s = 0.f, ss = 0.f;
    for (int i = tid; i < total4; i += 256) {
        float4 v = xp[i];
        s  += v.x + v.y + v.z + v.w;
        ss += v.x * v.x + v.y * v.y + v.z * v.z + v.w * v.w;
    }

    __shared__ float sh_s[256];
    __shared__ float sh_q[256];
    sh_s[tid] = s; sh_q[tid] = ss;
    __syncthreads();
    for (int st = 128; st > 0; st >>= 1) {
        if (tid < st) { sh_s[tid] += sh_s[tid + st]; sh_q[tid] += sh_q[tid + st]; }
        __syncthreads();
    }
    if (tid == 0) {
        const float inv = 1.0f / (float)(CIN * HW);
        float mu  = sh_s[0] * inv;
        float var = sh_q[0] * inv - mu * mu;
        stats[n]      = mu;
        stats[32 + n] = __frsqrt_rn(var + 1e-5f);
    }
}

// ---------------------------------------------------------------------------
// Kernel 2: s = mean(|w|)  (single block)
// ---------------------------------------------------------------------------
__global__ void wscale_kernel(const float* __restrict__ w, float* __restrict__ sout) {
    const int tid = threadIdx.x;
    float s = 0.f;
    for (int i = tid; i < NW; i += 256) s += fabsf(w[i]);
    __shared__ float sh[256];
    sh[tid] = s;
    __syncthreads();
    for (int st = 128; st > 0; st >>= 1) {
        if (tid < st) sh[tid] += sh[tid + st];
        __syncthreads();
    }
    if (tid == 0) sout[0] = sh[0] / (float)NW;
}

// ---------------------------------------------------------------------------
// Kernel 3: ternary quantize -> f16, layout [Cout][ off*64 + ci ]
// (tap-major K ordering so a 32-wide GEMM K-step has a constant filter tap)
// ---------------------------------------------------------------------------
__global__ void wquant_kernel(const float* __restrict__ w,
                              const float* __restrict__ sptr,
                              _Float16* __restrict__ wq) {
    const int idx = blockIdx.x * 256 + threadIdx.x;
    if (idx < NW) {
        const float s = sptr[0];
        float q = rintf(w[idx] / (s + 1e-8f));          // round-half-even like jnp.round
        q = fminf(1.0f, fmaxf(-1.0f, q));
        const int co  = idx / 576;
        const int rem = idx - co * 576;                 // ci*9 + off
        const int ci  = rem / 9;
        const int off = rem - ci * 9;
        wq[co * KTOT + off * 64 + ci] = (_Float16)q;
    }
}

// ---------------------------------------------------------------------------
// Kernel 4: fused GroupNorm + implicit-GEMM conv on WMMA.
// Block tile: 128 Cout x 128 pixels; 8 waves; wave w owns Cout rows
// [16w,16w+16) x eight 16-pixel N tiles -> 8 WMMAs per 32-wide K-step.
// K = 576 in 18 steps; double-buffered LDS; weight tile staged with async
// global->LDS (runs under the WMMAs), activations staged through registers
// with the GroupNorm fused as one FMA; single barrier per K-step.
// ---------------------------------------------------------------------------
__global__ __launch_bounds__(256)
void conv_gemm_kernel(const float* __restrict__ x,
                      const float* __restrict__ stats,
                      const _Float16* __restrict__ wq,
                      const float* __restrict__ gamma,
                      const float* __restrict__ beta,
                      const float* __restrict__ bias,
                      float* __restrict__ out) {
    __shared__ __align__(16) _Float16 A_lds[2][COUT * 32];   // [co][k] per buffer
    __shared__ __align__(16) _Float16 B_lds[2][NBLK * 32];   // [pix][k] per buffer
    __shared__ __align__(16) float nsc[64];                  // rstd*gamma[ci]
    __shared__ __align__(16) float nsh[64];                  // beta[ci]-mu*rstd*gamma[ci]

    const int tid  = threadIdx.x;
    const int lane = tid & 31;
    const int wv   = tid >> 5;            // 0..7
    const int row  = lane & 15;
    const int hi16 = lane >> 4;           // 0 or 1

    const int tile = blockIdx.x;          // 0..3135
    const int n    = tile / (HW / NBLK);  // 12544 % 128 == 0 -> tile within one sample
    const int p0   = (tile % (HW / NBLK)) * NBLK;

    const float mu   = stats[n];
    const float rstd = stats[32 + n];
    const float* xN  = x + (long)n * CIN * HW;

    // B staging mapping: pix = tid&127 (coalesced along W), kq = tid>>7 (0..1)
    const int pix_l = tid & 127;
    const int kq    = tid >> 7;
    const int p_my  = p0 + pix_l;
    const int h     = p_my / W_;
    const int w     = p_my - h * W_;

    // A staging mapping: co = tid>>1, half select = tid&1 (16 halves = 32B each)
    const int a_co   = tid >> 1;
    const int a_half = tid & 1;

    // ---- staging helpers -------------------------------------------------
    // Weights: raw copy -> async global->LDS (b128 x2, IOFFSET applies to both sides).
    auto stageA = [&](int kstep, _Float16* Ab) {
        _Float16*       dst = &Ab[a_co * 32 + a_half * 16];
        const _Float16* src = wq + a_co * KTOT + kstep + a_half * 16;
#if HAVE_ASYNC_LDS
        __builtin_amdgcn_global_load_async_to_lds_b128(
            (as1_v4i*)src, (as3_v4i*)dst, 0, 0);
        __builtin_amdgcn_global_load_async_to_lds_b128(
            (as1_v4i*)src, (as3_v4i*)dst, 16, 0);
#else
        ((v8h*)dst)[0] = ((const v8h*)src)[0];
        ((v8h*)dst)[1] = ((const v8h*)src)[1];
#endif
    };
    // tap-major K: off = kstep>>6 constant per K-step -> one border check and
    // one base address per thread per step; 16 b32 loads at immediate offsets.
    auto stageB_regs = [&](int kstep, float xv[16]) {
        const int off = kstep >> 6;               // filter tap 0..8
        const int r   = off / 3;
        const int s_  = off - 3 * r;
        const int hi  = h + r - 1;
        const int wi  = w + s_ - 1;
        const int ci0 = (kstep & 32) + kq * 16;   // 16-aligned channel base
        const bool valid = ((unsigned)hi < (unsigned)H_) && ((unsigned)wi < (unsigned)W_);
        const float* bp = xN + ci0 * HW + hi * W_ + wi;
#pragma unroll
        for (int kk = 0; kk < 16; ++kk)
            xv[kk] = valid ? bp[kk * HW] : 0.0f;
    };
    auto stageB_store = [&](int kstep, const float xv[16], _Float16* Bb) {
        const int ci0 = (kstep & 32) + kq * 16;
        __attribute__((aligned(16))) float scv[16];
        __attribute__((aligned(16))) float shv[16];
#pragma unroll
        for (int q4 = 0; q4 < 4; ++q4) {
            *(float4*)&scv[q4 * 4] = *(const float4*)&nsc[ci0 + q4 * 4];
            *(float4*)&shv[q4 * 4] = *(const float4*)&nsh[ci0 + q4 * 4];
        }
        __attribute__((aligned(16))) _Float16 tmp[16];
#pragma unroll
        for (int kk = 0; kk < 16; ++kk)
            tmp[kk] = (_Float16)fmaf(xv[kk], scv[kk], shv[kk]);
        *(v8h*)&Bb[pix_l * 32 + kq * 16]     = *(const v8h*)&tmp[0];
        *(v8h*)&Bb[pix_l * 32 + kq * 16 + 8] = *(const v8h*)&tmp[8];
    };

    // ---- prologue: norm tables + stage K-step 0 --------------------------
    if (tid < 64) {
        const float g  = gamma[tid];
        const float sc = rstd * g;
        nsc[tid] = sc;
        nsh[tid] = beta[tid] - mu * sc;
    }
    stageA(0, &A_lds[0][0]);                       // async copy of weight tile 0
    float pxv[16];
    stageB_regs(0, pxv);
    __syncthreads();                               // tables visible
    stageB_store(0, pxv, &B_lds[0][0]);
    wait_async_lds();                              // weight tile 0 landed
    __syncthreads();                               // tile 0 ready

    v8f acc[8] = {};

#pragma unroll 2
    for (int ks = 0; ks < NKS; ++ks) {
        const int kstep = ks * 32;
        const _Float16* Ab = &A_lds[ks & 1][0];
        const _Float16* Bb = &B_lds[ks & 1][0];

        // kick off next K-step: async weight copy + activation loads to regs
        float nxv[16];
        if (ks < NKS - 1) {
            stageA(kstep + 32, &A_lds[(ks + 1) & 1][0]);
            stageB_regs(kstep + 32, nxv);
        }

        // A fragment: per lane, halves j -> K = hi16*16 + j (contiguous, matches B)
        const int co_r = (wv * 16 + row) * 32;
        const v8h a_lo = *(const v8h*)&Ab[co_r + hi16 * 16];
        const v8h a_hi = *(const v8h*)&Ab[co_r + hi16 * 16 + 8];
        const v16h a = __builtin_shufflevector(a_lo, a_hi,
                           0, 1, 2, 3, 4, 5, 6, 7, 8, 9, 10, 11, 12, 13, 14, 15);

        // B fragments: per lane, halves j -> K = hi16*16 + j (contiguous)
#pragma unroll
        for (int nt = 0; nt < 8; ++nt) {
            const int pix = nt * 16 + row;
            const v8h b_lo = *(const v8h*)&Bb[pix * 32 + hi16 * 16];
            const v8h b_hi = *(const v8h*)&Bb[pix * 32 + hi16 * 16 + 8];
            const v16h b = __builtin_shufflevector(b_lo, b_hi,
                               0, 1, 2, 3, 4, 5, 6, 7, 8, 9, 10, 11, 12, 13, 14, 15);
            acc[nt] = __builtin_amdgcn_wmma_f32_16x16x32_f16(
                false, a, false, b, (short)0, acc[nt], false, false);
        }

        // normalize + commit next activation tile; drain async; one barrier
        if (ks < NKS - 1)
            stageB_store(kstep + 32, nxv, &B_lds[(ks + 1) & 1][0]);
        wait_async_lds();
        __syncthreads();
    }

    // ---- epilogue: C/D layout VGPR r: lanes0-15 M=r, lanes16-31 M=r+8 ----
    const int co_b = wv * 16;
#pragma unroll
    for (int nt = 0; nt < 8; ++nt) {
        const int p_out = p0 + nt * 16 + row;
#pragma unroll
        for (int r = 0; r < 8; ++r) {
            const int co = co_b + hi16 * 8 + r;
            out[(long)(n * COUT + co) * HW + p_out] = acc[nt][r] + bias[co];
        }
    }
}

// ---------------------------------------------------------------------------
// Launch. ws layout (floats): [0..31]=mu, [32..63]=rstd, [64]=scale,
// f16 quantized weights start at float offset 128 (147456 bytes).
// ---------------------------------------------------------------------------
extern "C" void kernel_launch(void* const* d_in, const int* in_sizes, int n_in,
                              void* d_out, int out_size, void* d_ws, size_t ws_size,
                              hipStream_t stream) {
    const float* x      = (const float*)d_in[0];
    const float* weight = (const float*)d_in[1];
    const float* bias   = (const float*)d_in[2];
    const float* gamma  = (const float*)d_in[3];
    const float* beta   = (const float*)d_in[4];
    float* out = (float*)d_out;

    float*    ws_f    = (float*)d_ws;
    float*    stats   = ws_f;            // 64 floats (mu, rstd)
    float*    s_scale = ws_f + 64;       // 1 float
    _Float16* wqp     = (_Float16*)(ws_f + 128);

    stats_kernel <<<32, 256, 0, stream>>>(x, stats);
    wscale_kernel<<<1, 256, 0, stream>>>(weight, s_scale);
    wquant_kernel<<<(NW + 255) / 256, 256, 0, stream>>>(weight, s_scale, wqp);

    conv_gemm_kernel<<<NPIX / NBLK, 256, 0, stream>>>(x, stats, wqp, gamma, beta, bias, out);
}
